// IDS4TokenClassifier_35974646071830
// MI455X (gfx1250) — matrix operand order repaired and checked by reading, not compiled, not run on hardware
//
#include <hip/hip_runtime.h>
#include <hip/hip_bf16.h>

// ---------------- problem dims (compile-time constants from reference) ------
#define DM     256                 // d_model
#define DS     64                  // d_state
#define NL     4                   // layers
#define NV     8192                // vocab
#define BATCH  4
#define SEQ    2048
#define BL     (BATCH * SEQ)       // 8192 token positions
#define DS2    (DS * DS)           // 4096 (one 64x64 state matrix)
#define CHUNK  64                  // scan chunk length
#define NCHUNK (SEQ / CHUNK)       // 32 chunks per batch

// ---------------- WMMA types ------------------------------------------------
typedef __attribute__((ext_vector_type(16))) __bf16 v16bf;
typedef __attribute__((ext_vector_type(4)))  __bf16 v4bf;
typedef __attribute__((ext_vector_type(8)))  float  v8f;

__device__ __forceinline__ __bf16 f2bf(float f) { return (__bf16)f; }

__device__ __forceinline__ v4bf pack4(float4 v) {
  v4bf p;
  p[0] = f2bf(v.x); p[1] = f2bf(v.y); p[2] = f2bf(v.z); p[3] = f2bf(v.w);
  return p;
}

// Fragment-order LDS swizzle (this code's contiguous-k order, same for A and B):
//  A 16x32 tile, element (r,k):  lane = (r&15) + 16*((k&31)>>4)
//                                e    = k & 15
//  B 32x16 tile, element (k,n):  lane = (n&15) + 16*((k&31)>>4)
//                                e    = k & 15
// LDS stores a tile-block as [lane][16] (512 bf16 = 1 KB); a lane's operand is
// 16 contiguous bf16 = 32 B (2x ds_load_b128, no packing movs).

__device__ __forceinline__ v8f wmma_bf16(v16bf a, v16bf b, v8f c) {
  // (neg_a, A, neg_b, B, c_mod, C, reuse_a, reuse_b)
  return __builtin_amdgcn_wmma_f32_16x16x32_bf16(false, a, false, b, (short)0, c,
                                                 false, false);
}

// ---------------- 256-thread (8-wave) block GEMM: 64x64 tile ----------------
// acc[2]: each wave owns sub-tiles (wave) and (wave+8) of the 4x4 16x16 grid.
// A is MxK f32 row-major (lda); B is KxN f32 row-major (ldb) or, if transB,
// NxK row-major (ldb = K-stride).  Demotes to bf16 in LDS, f32 accumulate.
// sAf/sBf: 2048 bf16 each, fragment order [block(4)][lane(32)][e(16)].
__device__ __forceinline__ void block_gemm64(const float* A, int lda,
                                             const float* Bm, int ldb, bool transB,
                                             int K, long row0, long col0,
                                             __bf16* sAf, __bf16* sBf, v8f acc[2]) {
  const int tid  = threadIdx.x;
  const int lane = tid & 31;
  const int wave = tid >> 5;
  for (int k0 = 0; k0 < K; k0 += 32) {
    if (k0 + 32 < K) {  // speculative prefetch of next A K-slice -> global_prefetch
      __builtin_prefetch(A + (row0 + (tid & 63)) * (long)lda + k0 + 32, 0, 0);
    }
    // ---- stage A 64x32 slice: 512 float4 groups, fully vectorized both ends
#pragma unroll
    for (int g = tid; g < 512; g += 256) {
      const int r = g >> 3, c4 = (g & 7) << 2;
      const float4 v = *(const float4*)(A + (row0 + r) * (long)lda + k0 + c4);
      const int ln = (r & 15) + ((c4 >> 4) << 4);
      const int e  = c4 & 15;
      *(v4bf*)(sAf + ((r >> 4) << 9) + (ln << 4) + e) = pack4(v);
    }
    // ---- stage B 32x64 slice
    if (transB) {  // W is NxK row-major: contiguous over k -> vectorized
#pragma unroll
      for (int g = tid; g < 512; g += 256) {
        const int c = g >> 3, r4 = (g & 7) << 2;
        const float4 v = *(const float4*)(Bm + (col0 + c) * (long)ldb + k0 + r4);
        const int ln = (c & 15) + ((r4 >> 4) << 4);
        *(v4bf*)(sBf + ((c >> 4) << 9) + (ln << 4) + (r4 & 15)) = pack4(v);
      }
    } else {       // B KxN row-major: vector global load, scattered b16 stores
#pragma unroll
      for (int g = tid; g < 512; g += 256) {
        const int r = g >> 4, c4 = (g & 15) << 2;
        const float4 v = *(const float4*)(Bm + (long)(k0 + r) * ldb + col0 + c4);
        const float vv[4] = {v.x, v.y, v.z, v.w};
#pragma unroll
        for (int j = 0; j < 4; ++j) {
          const int c = c4 + j;
          sBf[((c >> 4) << 9) + (((c & 15) + ((r >> 4) << 4)) << 4) + (r & 15)] =
              f2bf(vv[j]);
        }
      }
    }
    __syncthreads();
#pragma unroll
    for (int t = 0; t < 2; ++t) {
      const int sub = wave + t * 8;
      const v16bf fa = *(const v16bf*)(sAf + ((sub >> 2) << 9) + (lane << 4));
      const v16bf fb = *(const v16bf*)(sBf + ((sub & 3) << 9) + (lane << 4));
      acc[t] = wmma_bf16(fa, fb, acc[t]);
    }
    __syncthreads();
  }
}

// ---------------- kernel 1: embedding gather --------------------------------
__global__ void embed_kernel(const int* __restrict__ x,
                             const float* __restrict__ emb,
                             float* __restrict__ h) {
  const long i = (long)blockIdx.x * blockDim.x + threadIdx.x;  // over BL*DM
  const int tok = x[i >> 8];
  h[i] = emb[(long)tok * DM + (i & (DM - 1))];
}

// ---------------- kernel 2: Ax = relu(h @ A)  (M=BL, K=DM, N=DS2) -----------
__global__ __launch_bounds__(256) void ax_gemm_relu_kernel(
    const float* __restrict__ h, const float* __restrict__ Amat,
    float* __restrict__ Q) {
  __shared__ alignas(32) __bf16 sAf[2048];
  __shared__ alignas(32) __bf16 sBf[2048];
  v8f acc[2] = {};
  const long row0 = (long)blockIdx.y * 64, col0 = (long)blockIdx.x * 64;
  block_gemm64(h, DM, Amat, DS2, false, DM, row0, col0, sAf, sBf, acc);
  const int lane = threadIdx.x & 31, wave = threadIdx.x >> 5;
  const int n = lane & 15, mb = (lane >> 4) << 3;
#pragma unroll
  for (int t = 0; t < 2; ++t) {
    const int sub = wave + t * 8, sr = (sub >> 2) << 4, sc = (sub & 3) << 4;
#pragma unroll
    for (int i = 0; i < 8; ++i) {
      float v = acc[t][i];
      v = v > 0.f ? v : 0.f;                       // ReLU
      Q[(row0 + sr + mb + i) * (long)DS2 + col0 + sc + n] = v;
    }
  }
}

// ---------------- kernel 3: phase-1 scan — per-chunk serial prefix ----------
// In-place: Q[l] becomes Q_local[l] = Ax_{c0} @ ... @ Ax_l within its chunk.
// Running 64x64 product lives in LDS in A-fragment order sQf[rb(4)][ks(2)][lane][16].
__global__ __launch_bounds__(512) void scan_chunk_kernel(float* __restrict__ Q) {
  __shared__ alignas(32) __bf16 sQf[DS2];
  __shared__ alignas(32) __bf16 sXf[DS2];   // B-fragment order [cb][ks][lane][16]
  const int tid = threadIdx.x, lane = tid & 31, wave = tid >> 5;  // wave 0..15
  const int rb = wave >> 2, cb = wave & 3;
  const long base = (long)blockIdx.x * CHUNK;
  const float* x0 = Q + base * DS2;
  // stage initial product (P_1 = Ax_1) in A-fragment order, vectorized
#pragma unroll
  for (int g = tid; g < 1024; g += 512) {
    const int r = g >> 4, k4 = (g & 15) << 2;
    const float4 v = *(const float4*)(x0 + r * 64 + k4);
    const int kk = k4 & 31;
    const int ln = (r & 15) + ((kk >> 4) << 4);
    const int e  = kk & 15;
    *(v4bf*)(sQf + ((r >> 4) << 10) + ((k4 >> 5) << 9) + (ln << 4) + e) = pack4(v);
  }
  __syncthreads();
  for (int s = 1; s < CHUNK; ++s) {
    float* xs = Q + (base + s) * (long)DS2;
    // stage Ax_s in B-fragment order (vector global loads, b16 scatter stores)
#pragma unroll
    for (int g = tid; g < 1024; g += 512) {
      const int k = g >> 4, c4 = (g & 15) << 2;
      const float4 v = *(const float4*)(xs + k * 64 + c4);
      const float vv[4] = {v.x, v.y, v.z, v.w};
#pragma unroll
      for (int j = 0; j < 4; ++j) {
        const int c = c4 + j;
        sXf[((c >> 4) << 10) + ((k >> 5) << 9) +
            (((c & 15) + (((k & 31) >> 4) << 4)) << 4) + (k & 15)] = f2bf(vv[j]);
      }
    }
    __syncthreads();
    v8f acc = {};
#pragma unroll
    for (int ks = 0; ks < 2; ++ks) {                     // Q = Q @ Ax_s
      const v16bf fa = *(const v16bf*)(sQf + (rb << 10) + (ks << 9) + (lane << 4));
      const v16bf fb = *(const v16bf*)(sXf + (cb << 10) + (ks << 9) + (lane << 4));
      acc = wmma_bf16(fa, fb, acc);
    }
    __syncthreads();
    const int n = lane & 15, mb = (lane >> 4) << 3;
    const int sr = rb << 4, sc = cb << 4;
#pragma unroll
    for (int i = 0; i < 8; ++i) {
      const float vv = acc[i];
      const int r = sr + mb + i, c = sc + n;
      xs[r * 64 + c] = vv;            // global f32 prefix (overwrites Ax_s)
      const int kk = c & 31;          // scatter-update running product (A-order)
      const int ln = (r & 15) + ((kk >> 4) << 4);
      const int e  = kk & 15;
      sQf[((r >> 4) << 10) + ((c >> 5) << 9) + (ln << 4) + e] = f2bf(vv);
    }
    __syncthreads();
  }
}

// ---------------- kernel 4: phase-2 — scan of chunk totals per batch --------
__global__ __launch_bounds__(512) void chunk_prefix_kernel(
    const float* __restrict__ Q, float* __restrict__ Off) {
  __shared__ alignas(32) __bf16 sQf[DS2];
  __shared__ alignas(32) __bf16 sXf[DS2];
  const int tid = threadIdx.x, lane = tid & 31, wave = tid >> 5;
  const int rb = wave >> 2, cb = wave & 3;
  const int b = blockIdx.x;
  const float* t0 = Q + (long)(b * SEQ + CHUNK - 1) * DS2;       // chunk-0 total
  float* o0 = Off + (long)(b * NCHUNK) * DS2;
#pragma unroll
  for (int g = tid; g < 1024; g += 512) {
    const int r = g >> 4, k4 = (g & 15) << 2;
    const float4 v = *(const float4*)(t0 + r * 64 + k4);
    *(float4*)(o0 + r * 64 + k4) = v;
    const int kk = k4 & 31;
    const int ln = (r & 15) + ((kk >> 4) << 4);
    const int e  = kk & 15;
    *(v4bf*)(sQf + ((r >> 4) << 10) + ((k4 >> 5) << 9) + (ln << 4) + e) = pack4(v);
  }
  __syncthreads();
  for (int c2 = 1; c2 < NCHUNK; ++c2) {
    const float* tc = Q + (long)(b * SEQ + c2 * CHUNK + CHUNK - 1) * DS2;
    float* oc = Off + (long)(b * NCHUNK + c2) * DS2;
#pragma unroll
    for (int g = tid; g < 1024; g += 512) {
      const int k = g >> 4, c4 = (g & 15) << 2;
      const float4 v = *(const float4*)(tc + k * 64 + c4);
      const float vv[4] = {v.x, v.y, v.z, v.w};
#pragma unroll
      for (int j = 0; j < 4; ++j) {
        const int c = c4 + j;
        sXf[((c >> 4) << 10) + ((k >> 5) << 9) +
            (((c & 15) + (((k & 31) >> 4) << 4)) << 4) + (k & 15)] = f2bf(vv[j]);
      }
    }
    __syncthreads();
    v8f acc = {};
#pragma unroll
    for (int ks = 0; ks < 2; ++ks) {                    // O = O @ T_c
      const v16bf fa = *(const v16bf*)(sQf + (rb << 10) + (ks << 9) + (lane << 4));
      const v16bf fb = *(const v16bf*)(sXf + (cb << 10) + (ks << 9) + (lane << 4));
      acc = wmma_bf16(fa, fb, acc);
    }
    __syncthreads();
    const int n = lane & 15, mb = (lane >> 4) << 3;
    const int sr = rb << 4, sc = cb << 4;
#pragma unroll
    for (int i = 0; i < 8; ++i) {
      const float vv = acc[i];
      const int r = sr + mb + i, c = sc + n;
      oc[r * 64 + c] = vv;
      const int kk = c & 31;
      const int ln = (r & 15) + ((kk >> 4) << 4);
      const int e  = kk & 15;
      sQf[((r >> 4) << 10) + ((c >> 5) << 9) + (ln << 4) + e] = f2bf(vv);
    }
    __syncthreads();
  }
}

// ---------------- kernel 5: phase-3 fused with proj -------------------------
// cum_l = O_{c-1} @ (Q_local_l @ w) + pb   (no global prefix matrices needed)
__global__ __launch_bounds__(64) void cum_kernel(
    const float* __restrict__ Q, const float* __restrict__ Off,
    const float* __restrict__ pw, const float* __restrict__ pb,
    float* __restrict__ cum) {
  const int l = blockIdx.x;           // 0..BL-1
  const int s = threadIdx.x;          // 0..63
  const int b = l / SEQ, lb = l % SEQ, c = lb / CHUNK;
  __shared__ float v[DS];
  const float* P = Q + (long)l * DS2;
  float acc = 0.f;
#pragma unroll 4
  for (int t = 0; t < DS; ++t) acc += P[s * DS + t] * pw[t];
  v[s] = acc;
  __syncthreads();
  float outv;
  if (c == 0) {
    outv = v[s];
  } else {
    const float* O = Off + (long)(b * NCHUNK + c - 1) * DS2;
    float a2 = 0.f;
#pragma unroll 4
    for (int t = 0; t < DS; ++t) a2 += O[s * DS + t] * v[t];
    outv = a2;
  }
  cum[(long)l * DS + s] = outv + pb[0];
}

// ---------------- kernel 6: z + residual: hz = h + cum@Cw^T + h@Dw^T + b ----
__global__ __launch_bounds__(256) void mix_kernel(
    const float* __restrict__ h, const float* __restrict__ cum,
    const float* __restrict__ Cw, const float* __restrict__ Cb,
    const float* __restrict__ Dw, const float* __restrict__ Db,
    float* __restrict__ hz) {
  __shared__ alignas(32) __bf16 sAf[2048];
  __shared__ alignas(32) __bf16 sBf[2048];
  v8f acc[2] = {};
  const long row0 = (long)blockIdx.y * 64, col0 = (long)blockIdx.x * 64;
  block_gemm64(h, DM, Dw, DM, true, DM, row0, col0, sAf, sBf, acc);   // h  @ Dw^T
  block_gemm64(cum, DS, Cw, DS, true, DS, row0, col0, sAf, sBf, acc); // cum@ Cw^T
  const int lane = threadIdx.x & 31, wave = threadIdx.x >> 5;
  const int n = lane & 15, mb = (lane >> 4) << 3;
#pragma unroll
  for (int t = 0; t < 2; ++t) {
    const int sub = wave + t * 8, sr = (sub >> 2) << 4, sc = (sub & 3) << 4;
#pragma unroll
    for (int i = 0; i < 8; ++i) {
      const long r = row0 + sr + mb + i;
      const int  cc = (int)(col0 + sc + n);
      hz[r * DM + cc] = acc[t][i] + Cb[cc] + Db[cc] + h[r * DM + cc];
    }
  }
}

// ---------------- kernel 7: LayerNorm over DM=256, one row per block --------
__global__ __launch_bounds__(256) void ln_kernel(const float* __restrict__ hz,
                                                 const float* __restrict__ g,
                                                 const float* __restrict__ bb,
                                                 float* __restrict__ h) {
  const int row = blockIdx.x, t = threadIdx.x;
  __shared__ float red[256];
  const float x = hz[(long)row * DM + t];
  red[t] = x;
  __syncthreads();
  for (int s = 128; s > 0; s >>= 1) {
    if (t < s) red[t] += red[t + s];
    __syncthreads();
  }
  const float mu = red[0] * (1.0f / DM);
  __syncthreads();
  const float d = x - mu;
  red[t] = d * d;
  __syncthreads();
  for (int s = 128; s > 0; s >>= 1) {
    if (t < s) red[t] += red[t + s];
    __syncthreads();
  }
  const float var = red[0] * (1.0f / DM);
  h[(long)row * DM + t] = d * rsqrtf(var + 1e-5f) * g[t] + bb[t];
}

// ---------------- kernel 8: head: out = h @ head_w^T + head_b ---------------
__global__ __launch_bounds__(256) void head_kernel(
    const float* __restrict__ h, const float* __restrict__ W,
    const float* __restrict__ bias, float* __restrict__ out) {
  __shared__ alignas(32) __bf16 sAf[2048];
  __shared__ alignas(32) __bf16 sBf[2048];
  v8f acc[2] = {};
  const long row0 = (long)blockIdx.y * 64, col0 = (long)blockIdx.x * 64;
  block_gemm64(h, DM, W, DM, true, DM, row0, col0, sAf, sBf, acc);
  const int lane = threadIdx.x & 31, wave = threadIdx.x >> 5;
  const int n = lane & 15, mb = (lane >> 4) << 3;
#pragma unroll
  for (int t = 0; t < 2; ++t) {
    const int sub = wave + t * 8, sr = (sub >> 2) << 4, sc = (sub & 3) << 4;
#pragma unroll
    for (int i = 0; i < 8; ++i) {
      const long r = row0 + sr + mb + i;
      const long cc = col0 + sc + n;
      out[r * (long)NV + cc] = acc[t][i] + bias[cc];
    }
  }
}

// ---------------- host launch ----------------------------------------------
extern "C" void kernel_launch(void* const* d_in, const int* in_sizes, int n_in,
                              void* d_out, int out_size, void* d_ws, size_t ws_size,
                              hipStream_t stream) {
  (void)in_sizes; (void)n_in; (void)out_size; (void)ws_size;
  const int*   x      = (const int*)d_in[0];
  const float* emb    = (const float*)d_in[1];
  const float* head_w = (const float*)d_in[2];
  const float* head_b = (const float*)d_in[3];

  // workspace partition (~148 MB total)
  char* ws = (char*)d_ws;
  float* h   = (float*)ws; ws += (size_t)BL * DM * sizeof(float);              // 8 MB
  float* Q   = (float*)ws; ws += (size_t)BL * DS2 * sizeof(float);             // 128 MB
  float* Off = (float*)ws; ws += (size_t)BATCH * NCHUNK * DS2 * sizeof(float); // 2 MB
  float* cum = (float*)ws; ws += (size_t)BL * DS * sizeof(float);              // 2 MB
  float* hz  = (float*)ws;                                                     // 8 MB

  embed_kernel<<<(BL * DM) / 256, 256, 0, stream>>>(x, emb, h);

  for (int li = 0; li < NL; ++li) {
    const int p = 4 + li * 9;  // A, proj_w, proj_b, C_w, C_b, D_w, D_b, ln_g, ln_b
    const float* Amat = (const float*)d_in[p + 0];
    const float* pw   = (const float*)d_in[p + 1];
    const float* pb   = (const float*)d_in[p + 2];
    const float* Cw   = (const float*)d_in[p + 3];
    const float* Cb   = (const float*)d_in[p + 4];
    const float* Dw   = (const float*)d_in[p + 5];
    const float* Db   = (const float*)d_in[p + 6];
    const float* lng  = (const float*)d_in[p + 7];
    const float* lnb  = (const float*)d_in[p + 8];

    ax_gemm_relu_kernel<<<dim3(DS2 / 64, BL / 64), 256, 0, stream>>>(h, Amat, Q);
    scan_chunk_kernel<<<BATCH * NCHUNK, 512, 0, stream>>>(Q);
    chunk_prefix_kernel<<<BATCH, 512, 0, stream>>>(Q, Off);
    cum_kernel<<<BL, 64, 0, stream>>>(Q, Off, pw, pb, cum);
    mix_kernel<<<dim3(DM / 64, BL / 64), 256, 0, stream>>>(h, cum, Cw, Cb, Dw, Db, hz);
    ln_kernel<<<BL, 256, 0, stream>>>(hz, lng, lnb, h);
  }

  head_kernel<<<dim3(NV / 64, BL / 64), 256, 0, stream>>>(h, head_w, head_b,
                                                          (float*)d_out);
}
